// Decoder_19009525252565
// MI455X (gfx1250) — matrix operand order repaired and checked by reading, not compiled, not run on hardware
//
#include <hip/hip_runtime.h>
#include <hip/hip_bf16.h>

// ---- problem constants (match the reference) ----
#define B_      64
#define T_      1024
#define NMELS_  80
#define DEC_    128
#define LSTM_   512
#define CTX_    512
#define PRE_    256
#define FILT_   32
#define MAXR_   20

typedef __attribute__((ext_vector_type(16))) __bf16 v16bf;
typedef __attribute__((ext_vector_type(8)))  float  v8f;

__device__ __forceinline__ float sigm(float x) { return 1.0f / (1.0f + __expf(-x)); }

__device__ __forceinline__ v16bf pack_bf16(float4 a, float4 b, float4 c, float4 d)
{
  v16bf r;
  r[0]  = (__bf16)a.x; r[1]  = (__bf16)a.y; r[2]  = (__bf16)a.z; r[3]  = (__bf16)a.w;
  r[4]  = (__bf16)b.x; r[5]  = (__bf16)b.y; r[6]  = (__bf16)b.z; r[7]  = (__bf16)b.w;
  r[8]  = (__bf16)c.x; r[9]  = (__bf16)c.y; r[10] = (__bf16)c.z; r[11] = (__bf16)c.w;
  r[12] = (__bf16)d.x; r[13] = (__bf16)d.y; r[14] = (__bf16)d.z; r[15] = (__bf16)d.w;
  return r;
}

// ---------------------------------------------------------------------
// Shared WMMA K-loop: acc += A-row-fragment @ W-col-fragment over K.
// A fragment: lane = row, two contiguous 8-float runs -> 4x b128 loads.
// B fragment: lane owns column n = lane%16 (a row of W); 16 k-values
//             in the same two 8-float runs (table order) as the A fragment.
// ---------------------------------------------------------------------
template <int K>
__device__ __forceinline__ void gemm_accum(const float* __restrict__ arow,
                                           const float* __restrict__ wcol,
                                           v8f& acc)
{
  static_assert(K % 32 == 0, "K must be a multiple of 32");
#pragma unroll 2
  for (int k0 = 0; k0 < K; k0 += 32) {
    const float4 a0 = *(const float4*)(arow + k0);
    const float4 a1 = *(const float4*)(arow + k0 + 4);
    const float4 a2 = *(const float4*)(arow + k0 + 16);
    const float4 a3 = *(const float4*)(arow + k0 + 20);
    const v16bf afrag = pack_bf16(a0, a1, a2, a3);

    v16bf bfrag;
#pragma unroll
    for (int e = 0; e < 16; ++e)
      bfrag[e] = (__bf16)wcol[k0 + e + (e >= 8 ? 8 : 0)];   // k = table order, n = lane

    acc = __builtin_amdgcn_wmma_f32_16x16x32_bf16(false, afrag, false, bfrag,
                                                  (short)0, acc, false, false);
  }
}

// =====================================================================
// WMMA GEMM: C[M,N] = act(A[M,K] @ W[N,K]^T + bias)   (bf16 in, f32 acc)
// One wave -> one 16x16 C tile; 4 waves/block cover 64 N columns.
// =====================================================================
template <int ACT, int K>   // ACT: 0 = none, 1 = relu
__global__ void __launch_bounds__(128)
gemm_wmma_bf16(const float* __restrict__ A, const float* __restrict__ W,
               const float* __restrict__ bias, float* __restrict__ C, int N)
{
  const int lane = threadIdx.x & 31;
  const int wave = threadIdx.x >> 5;
  const int n0   = blockIdx.x * 64 + wave * 16;
  const int m0   = blockIdx.y * 16;
  const int half = lane >> 4;
  const int l15  = lane & 15;

  v8f acc = {};
  gemm_accum<K>(A + (size_t)(m0 + l15) * K + half * 8,
                W + (size_t)(n0 + l15) * K + half * 8, acc);

  // C layout: lane -> N (l15), reg r -> M row (+8 for upper half)
  const int   n  = n0 + l15;
  const float bv = bias ? bias[n] : 0.0f;
#pragma unroll
  for (int r = 0; r < 8; ++r) {
    float v = acc[r] + bv;
    if (ACT == 1) v = fmaxf(v, 0.0f);
    C[(size_t)(m0 + r + 8 * half) * N + n] = v;
  }
}

// =====================================================================
// Fused dual-input WMMA GEMM (LSTM gates):
//   C[M,N] = A1[M,K1] @ W1[N,K1]^T + A2[M,K2] @ W2[N,K2]^T + b1 + b2
// Both K-loops accumulate into the same WMMA accumulator registers —
// no intermediate gate buffers, half the launches.
// =====================================================================
template <int K1, int K2>
__global__ void __launch_bounds__(128)
gemm2_wmma_bf16(const float* __restrict__ A1, const float* __restrict__ W1,
                const float* __restrict__ b1,
                const float* __restrict__ A2, const float* __restrict__ W2,
                const float* __restrict__ b2,
                float* __restrict__ C, int N)
{
  const int lane = threadIdx.x & 31;
  const int wave = threadIdx.x >> 5;
  const int n0   = blockIdx.x * 64 + wave * 16;
  const int m0   = blockIdx.y * 16;
  const int half = lane >> 4;
  const int l15  = lane & 15;

  v8f acc = {};
  gemm_accum<K1>(A1 + (size_t)(m0 + l15) * K1 + half * 8,
                 W1 + (size_t)(n0 + l15) * K1 + half * 8, acc);
  gemm_accum<K2>(A2 + (size_t)(m0 + l15) * K2 + half * 8,
                 W2 + (size_t)(n0 + l15) * K2 + half * 8, acc);

  const int   n  = n0 + l15;
  const float bv = b1[n] + b2[n];
#pragma unroll
  for (int r = 0; r < 8; ++r)
    C[(size_t)(m0 + r + 8 * half) * N + n] = acc[r] + bv;
}

// ===================== elementwise / small kernels =====================

// zero-padded row copy: [rows, kin] -> [rows, kout]
__global__ void pad_rows(const float* __restrict__ src, float* __restrict__ dst,
                         int rows, int kin, int kout)
{
  int idx = blockIdx.x * blockDim.x + threadIdx.x;
  if (idx >= rows * kout) return;
  int r = idx / kout, c = idx % kout;
  dst[idx] = (c < kin) ? src[r * kin + c] : 0.0f;
}

__global__ void dropout_det(float* __restrict__ x, int n, unsigned seed)
{
  int i = blockIdx.x * blockDim.x + threadIdx.x;
  if (i >= n) return;
  unsigned h = (unsigned)i * 2654435761u ^ seed;
  h ^= h >> 16; h *= 0x7feb352du; h ^= h >> 15; h *= 0x846ca68bu; h ^= h >> 16;
  x[i] = (h & 1u) ? x[i] * 2.0f : 0.0f;     // keep=0.5, scale 1/keep
}

__global__ void concat2(const float* __restrict__ a, int na,
                        const float* __restrict__ b, int nb,
                        float* __restrict__ out, int rows)
{
  int idx = blockIdx.x * blockDim.x + threadIdx.x;
  int tot = na + nb;
  if (idx >= rows * tot) return;
  int r = idx / tot, c = idx % tot;
  out[idx] = (c < na) ? a[r * na + c] : b[r * nb + (c - na)];
}

__global__ void gru_combine(const float* __restrict__ gi, const float* __restrict__ gh,
                            const float* __restrict__ hprev, float* __restrict__ hout)
{
  int idx = blockIdx.x * blockDim.x + threadIdx.x;
  if (idx >= B_ * DEC_) return;
  int b = idx / DEC_, d = idx % DEC_;
  const float* gib = gi + (size_t)b * 3 * DEC_;
  const float* ghb = gh + (size_t)b * 3 * DEC_;
  float r  = sigm(gib[d] + ghb[d]);
  float z  = sigm(gib[DEC_ + d] + ghb[DEC_ + d]);
  float nn = tanhf(gib[2 * DEC_ + d] + r * ghb[2 * DEC_ + d]);
  hout[idx] = (1.0f - z) * nn + z * hprev[idx];
}

// g = fused (x@Wih^T + h@Whh^T + biases)  [B, 4*LSTM]; i,f,g,o split
__global__ void lstm_combine(const float* __restrict__ g,
                             const float* __restrict__ cprev, const float* __restrict__ xin,
                             float* __restrict__ hout, float* __restrict__ cout,
                             float* __restrict__ xout)
{
  int idx = blockIdx.x * blockDim.x + threadIdx.x;
  if (idx >= B_ * LSTM_) return;
  int b = idx / LSTM_, d = idx % LSTM_;
  const float* G = g + (size_t)b * 4 * LSTM_;
  float gi = G[d];
  float gf = G[LSTM_ + d];
  float gg = G[2 * LSTM_ + d];
  float go = G[3 * LSTM_ + d];
  float c2 = sigm(gf) * cprev[idx] + sigm(gi) * tanhf(gg);
  float h2 = sigm(go) * tanhf(c2);
  hout[idx] = h2;
  cout[idx] = c2;
  xout[idx] = xin[idx] + h2;       // residual
}

// pl[d] = sum_f conv_b[f] * L_w[d,f]   (conv over zero cumulative attention at t=0)
__global__ void ploc_const(const float* __restrict__ conv_b,
                           const float* __restrict__ L_w, float* __restrict__ pl)
{
  int d = threadIdx.x;
  if (d < DEC_) {
    float s = 0.f;
#pragma unroll
    for (int f = 0; f < FILT_; ++f) s += conv_b[f] * L_w[d * FILT_ + f];
    pl[d] = s;
  }
}

__global__ void attn_energy(const float* __restrict__ pq, const float* __restrict__ proj,
                            const float* __restrict__ pl, const float* __restrict__ vw,
                            const int* __restrict__ chars, float* __restrict__ u)
{
  int idx = blockIdx.x * blockDim.x + threadIdx.x;
  if (idx >= B_ * T_) return;
  int b = idx / T_;
  const float4* pr = (const float4*)(proj + (size_t)idx * DEC_);
  const float4* pb = (const float4*)(pq + (size_t)b * DEC_);
  const float4* pl4 = (const float4*)pl;
  const float4* vw4 = (const float4*)vw;
  float s = 0.f;
#pragma unroll 4
  for (int q = 0; q < DEC_ / 4; ++q) {
    float4 x = pr[q], y = pb[q], z = pl4[q], v = vw4[q];
    s += v.x * tanhf(y.x + x.x + z.x);
    s += v.y * tanhf(y.y + x.y + z.y);
    s += v.z * tanhf(y.z + x.z + z.z);
    s += v.w * tanhf(y.w + x.w + z.w);
  }
  u[idx] = (chars[idx] != 0) ? s : 0.0f;
}

__global__ void softmax_rows(const float* __restrict__ u, float* __restrict__ s)
{
  __shared__ float red[8];
  const int b = blockIdx.x, tid = threadIdx.x;
  const float* ub = u + (size_t)b * T_;
  float* sb = s + (size_t)b * T_;

  float mx = -3.4e38f;
  for (int t = tid; t < T_; t += 256) mx = fmaxf(mx, ub[t]);
  for (int o = 16; o; o >>= 1) mx = fmaxf(mx, __shfl_down(mx, o, 32));
  if ((tid & 31) == 0) red[tid >> 5] = mx;
  __syncthreads();
  mx = red[0];
#pragma unroll
  for (int w = 1; w < 8; ++w) mx = fmaxf(mx, red[w]);

  float sum = 0.f;
  for (int t = tid; t < T_; t += 256) { float e = __expf(ub[t] - mx); sb[t] = e; sum += e; }
  for (int o = 16; o; o >>= 1) sum += __shfl_down(sum, o, 32);
  __syncthreads();
  if ((tid & 31) == 0) red[tid >> 5] = sum;
  __syncthreads();
  sum = 0.f;
#pragma unroll
  for (int w = 0; w < 8; ++w) sum += red[w];
  float inv = 1.0f / sum;
  for (int t = tid; t < T_; t += 256) sb[t] *= inv;
}

// context[b, col] = sum_t scores[b,t] * enc[b,t,col]   (~134 MB stream, BW-bound)
__global__ void context_reduce(const float* __restrict__ scores,
                               const float* __restrict__ enc,
                               float* __restrict__ out)
{
  __shared__ float sc[256];
  const int b   = blockIdx.y;
  const int col = blockIdx.x * 128 + threadIdx.x;
  const float* E = enc + (size_t)b * T_ * CTX_ + col;
  const float* S = scores + (size_t)b * T_;
  float acc = 0.f;
  for (int t0 = 0; t0 < T_; t0 += 256) {
    __syncthreads();
    sc[threadIdx.x]       = S[t0 + threadIdx.x];
    sc[threadIdx.x + 128] = S[t0 + threadIdx.x + 128];
    __syncthreads();
#pragma unroll 8
    for (int tt = 0; tt < 256; ++tt) {
      if ((tt & 63) == 0 && (t0 + tt + 64) < T_)
        __builtin_prefetch(E + (size_t)(t0 + tt + 64) * CTX_, 0, 1);  // global_prefetch
      acc += sc[tt] * E[(size_t)(t0 + tt) * CTX_];
    }
  }
  out[(size_t)b * CTX_ + col] = acc;
}

// mels[b, m, 0] = dot(x[b], mp_w[m*MAXR_])   (only r=0 survives the slice)
__global__ void mels_proj(const float* __restrict__ x, const float* __restrict__ mp_w,
                          float* __restrict__ mels)
{
  int idx = blockIdx.x * blockDim.x + threadIdx.x;
  if (idx >= B_ * NMELS_) return;
  int b = idx / NMELS_, mrow = idx % NMELS_;
  const float4* xr = (const float4*)(x + (size_t)b * LSTM_);
  const float4* w  = (const float4*)(mp_w + (size_t)(mrow * MAXR_) * LSTM_);
  float s = 0.f;
#pragma unroll 4
  for (int q = 0; q < LSTM_ / 4; ++q) {
    float4 a = xr[q], c = w[q];
    s += a.x * c.x + a.y * c.y + a.z * c.z + a.w * c.w;
  }
  mels[idx] = s;
}

__global__ void stop_proj(const float* __restrict__ x, const float* __restrict__ ctx,
                          const float* __restrict__ sp_w, const float* __restrict__ sp_b,
                          float* __restrict__ stop)
{
  int b = blockIdx.x, lane = threadIdx.x;   // one wave per batch row
  float s = 0.f;
  for (int k = lane; k < LSTM_; k += 32) s += x[(size_t)b * LSTM_ + k] * sp_w[k];
  for (int k = lane; k < CTX_; k += 32)  s += ctx[(size_t)b * CTX_ + k] * sp_w[LSTM_ + k];
  for (int o = 16; o; o >>= 1) s += __shfl_down(s, o, 32);
  if (lane == 0) stop[b] = sigm(s + sp_b[0]);
}

// =====================================================================
//                          host-side orchestration
// =====================================================================
template <int K>
static inline void launch_gemm(hipStream_t st, const float* A, const float* W,
                               const float* bias, float* C, int M, int N, int act)
{
  dim3 grid(N / 64, M / 16);
  if (act) gemm_wmma_bf16<1, K><<<grid, 128, 0, st>>>(A, W, bias, C, N);
  else     gemm_wmma_bf16<0, K><<<grid, 128, 0, st>>>(A, W, bias, C, N);
}

extern "C" void kernel_launch(void* const* d_in, const int* in_sizes, int n_in,
                              void* d_out, int out_size, void* d_ws, size_t ws_size,
                              hipStream_t stream)
{
  (void)in_sizes; (void)n_in; (void)out_size; (void)ws_size;

  const float* enc      = (const float*)d_in[0];   // [B,T,CTX]
  const float* proj     = (const float*)d_in[1];   // [B,T,DEC]
  const float* pre_in   = (const float*)d_in[2];   // [B,80]
  const float* attn_h0  = (const float*)d_in[3];   // [B,DEC]
  const float* r1_h     = (const float*)d_in[4];
  const float* r2_h     = (const float*)d_in[5];
  const float* r1_c     = (const float*)d_in[6];
  const float* r2_c     = (const float*)d_in[7];
  const float* ctx_vec  = (const float*)d_in[8];   // [B,CTX]
  const int*   chars    = (const int*)d_in[9];     // [B,T]
  // d_in[10] = t (==0), d_in[11] = conv_w: unused (conv over zeros)
  const float* conv_b   = (const float*)d_in[12];
  const float* L_w      = (const float*)d_in[13];
  const float* W_w      = (const float*)d_in[14];
  const float* W_b      = (const float*)d_in[15];
  const float* v_w      = (const float*)d_in[16];
  const float* fc1_w    = (const float*)d_in[17];
  const float* fc1_b    = (const float*)d_in[18];
  const float* fc2_w    = (const float*)d_in[19];
  const float* fc2_b    = (const float*)d_in[20];
  const float* gru_wih  = (const float*)d_in[21];
  const float* gru_whh  = (const float*)d_in[22];
  const float* gru_bih  = (const float*)d_in[23];
  const float* gru_bhh  = (const float*)d_in[24];
  const float* ri_w     = (const float*)d_in[25];
  const float* ri_b     = (const float*)d_in[26];
  const float* l1_wih   = (const float*)d_in[27];
  const float* l1_whh   = (const float*)d_in[28];
  const float* l1_bih   = (const float*)d_in[29];
  const float* l1_bhh   = (const float*)d_in[30];
  const float* l2_wih   = (const float*)d_in[31];
  const float* l2_whh   = (const float*)d_in[32];
  const float* l2_bih   = (const float*)d_in[33];
  const float* l2_bhh   = (const float*)d_in[34];
  const float* mp_w     = (const float*)d_in[35];
  const float* sp_w     = (const float*)d_in[36];
  const float* sp_b     = (const float*)d_in[37];

  // ---- output layout (flat, reference return order) ----
  float* out     = (float*)d_out;
  float* o_mels  = out;                 // 64*80*1   = 5120
  float* o_score = out + 5120;          // 64*1*1024 = 65536
  float* o_attnh = out + 70656;         // 64*128
  float* o_h1    = out + 78848;         // 64*512
  float* o_h2    = out + 111616;        // 64*512
  float* o_c1    = out + 144384;        // 64*512
  float* o_c2    = out + 177152;        // 64*512
  float* o_ctx   = out + 209920;        // 64*512
  float* o_stop  = out + 242688;        // 64

  // ---- workspace layout (floats; ~2 MB total) ----
  float* ws       = (float*)d_ws;
  float* w_pre1   = ws;                  // 64*256
  float* w_preo   = ws + 16384;          // 64*256
  float* w_gruin  = ws + 32768;          // 64*768
  float* w_gi     = ws + 81920;          // 64*384
  float* w_gh     = ws + 106496;         // 64*384
  float* w_pq     = ws + 131072;         // 64*128
  float* w_pl     = ws + 139264;         // 128
  float* w_u      = ws + 139392;         // 64*1024
  float* w_riin   = ws + 204928;         // 64*640
  float* w_x      = ws + 245888;         // 64*512
  float* w_g      = ws + 278656;         // 64*2048 (fused LSTM gates)
  float* w_x2     = ws + 409728;         // 64*512
  float* w_pin96  = ws + 442496;         // 64*96   (prenet_in padded K 80->96)
  float* w_fcw96  = ws + 448640;         // 256*96  (fc1_w padded K 80->96)

  // ---------------- PreNet (K padded 80 -> 96 so all GEMMs are guard-free) ----
  pad_rows<<<(B_ * 96 + 255) / 256, 256, 0, stream>>>(pre_in, w_pin96, B_, NMELS_, 96);
  pad_rows<<<(PRE_ * 96 + 255) / 256, 256, 0, stream>>>(fc1_w, w_fcw96, PRE_, NMELS_, 96);
  launch_gemm<96>(stream, w_pin96, w_fcw96, fc1_b, w_pre1, B_, PRE_, 1);
  dropout_det<<<64, 256, 0, stream>>>(w_pre1, B_ * PRE_, 0x9e3779b9u);
  launch_gemm<256>(stream, w_pre1, fc2_w, fc2_b, w_preo, B_, PRE_, 1);
  dropout_det<<<64, 256, 0, stream>>>(w_preo, B_ * PRE_, 0x85ebca6bu);

  // ---------------- attention GRU (gi / gh kept separate: r-gate couples them) ----
  concat2<<<192, 256, 0, stream>>>(ctx_vec, CTX_, w_preo, PRE_, w_gruin, B_);
  launch_gemm<768>(stream, w_gruin, gru_wih, gru_bih, w_gi, B_, 3 * DEC_, 0);
  launch_gemm<128>(stream, attn_h0, gru_whh, gru_bhh, w_gh, B_, 3 * DEC_, 0);
  gru_combine<<<32, 256, 0, stream>>>(w_gi, w_gh, attn_h0, o_attnh);

  // ---------------- LSA attention (t==0: conv term is constant) ----------------
  launch_gemm<128>(stream, o_attnh, W_w, W_b, w_pq, B_, DEC_, 0);
  ploc_const<<<1, 128, 0, stream>>>(conv_b, L_w, w_pl);
  attn_energy<<<(B_ * T_ + 255) / 256, 256, 0, stream>>>(w_pq, proj, w_pl, v_w, chars, w_u);
  softmax_rows<<<B_, 256, 0, stream>>>(w_u, o_score);
  context_reduce<<<dim3(CTX_ / 128, B_), 128, 0, stream>>>(o_score, enc, o_ctx);

  // ---------------- residual LSTM stack (fused gate GEMMs) ----------------
  concat2<<<160, 256, 0, stream>>>(o_ctx, CTX_, o_attnh, DEC_, w_riin, B_);
  launch_gemm<640>(stream, w_riin, ri_w, ri_b, w_x, B_, LSTM_, 0);

  {
    dim3 grid((4 * LSTM_) / 64, B_ / 16);
    gemm2_wmma_bf16<512, 512><<<grid, 128, 0, stream>>>(
        w_x, l1_wih, l1_bih, r1_h, l1_whh, l1_bhh, w_g, 4 * LSTM_);
  }
  lstm_combine<<<128, 256, 0, stream>>>(w_g, r1_c, w_x, o_h1, o_c1, w_x2);

  {
    dim3 grid((4 * LSTM_) / 64, B_ / 16);
    gemm2_wmma_bf16<512, 512><<<grid, 128, 0, stream>>>(
        w_x2, l2_wih, l2_bih, r2_h, l2_whh, l2_bhh, w_g, 4 * LSTM_);
  }
  lstm_combine<<<128, 256, 0, stream>>>(w_g, r2_c, w_x2, o_h2, o_c2, w_x);

  // ---------------- output projections ----------------
  mels_proj<<<(B_ * NMELS_ + 255) / 256, 256, 0, stream>>>(w_x, mp_w, o_mels);
  stop_proj<<<B_, 32, 0, stream>>>(w_x, o_ctx, sp_w, sp_b, o_stop);
}
